// RetNet_83253646066158
// MI455X (gfx1250) — matrix operand order repaired and checked by reading, not compiled, not run on hardware
//
#include <hip/hip_runtime.h>
#include <hip/hip_bf16.h>

// ---------------------------------------------------------------------------
// RetNet block for MI455X (gfx1250, wave32, WMMA).
// All GEMMs run through v_wmma_f32_16x16x32_bf16 (f32 accumulate).
// Staging uses CDNA5 async memory->LDS copies (ASYNCcnt) with double
// buffering, so the tile transfer overlaps the WMMA compute.
// ---------------------------------------------------------------------------

#define S_LEN   2048
#define DMODEL  1024
#define NHEADS  16
#define HEADD   64
#define HIDDEN  2730
#define HPAD    2752        // HIDDEN rounded to 64
#define NIN     5460        // 2*HIDDEN
#define NINPAD  5504        // rounded to 64
#define EPSV    1e-5f

typedef __attribute__((ext_vector_type(16))) __bf16 v16bf;
typedef __attribute__((ext_vector_type(8)))  float  v8f;

union AFrag { v16bf v; __bf16 e[16]; unsigned int u[8]; };
union BFrag { v16bf v; __bf16 e[16]; uint4 q[2]; };
union CFrag { v8f   v; float  e[8];  };

// 16-bit A-operand k index for element pair j (pos 0/1 are k, k+1),
// lane-half hi (ISA 7.12.2): lanes 0-15: V0..3 -> k 0..7, V4..7 -> k 16..23;
// lanes 16-31: +8.
__device__ __forceinline__ int a_kpair(int j, int hi) {
    return (j < 4 ? 2 * j : 8 + 2 * j) + 8 * hi;
}

// CDNA5 async memory -> LDS copy (tracked by ASYNCcnt, no VGPR bounce).
// LDS generic pointers carry the LDS byte offset in their low 32 bits.
__device__ __forceinline__ void async_copy_b128(void* lds, const void* g) {
    unsigned laddr = (unsigned)(unsigned long long)(size_t)lds;
    unsigned long long gaddr = (unsigned long long)(size_t)g;
    asm volatile("global_load_async_to_lds_b128 %0, %1, off"
                 :: "v"(laddr), "v"(gaddr) : "memory");
}
__device__ __forceinline__ void wait_async() {
    asm volatile("s_wait_asynccnt 0x0" ::: "memory");
}

// ---------------------------------------------------------------------------
// Fused LayerNorm -> bf16 activation
// ---------------------------------------------------------------------------
__global__ __launch_bounds__(256)
void k_ln(const float* __restrict__ x, const float* __restrict__ w,
          const float* __restrict__ b, __bf16* __restrict__ outb)
{
    __shared__ float s1[256], s2[256];
    const int s = blockIdx.x, tid = threadIdx.x;
    float vals[4], sum = 0.f, sq = 0.f;
    #pragma unroll
    for (int i = 0; i < 4; ++i) {
        float v = x[(size_t)s * DMODEL + tid + i * 256];
        vals[i] = v; sum += v; sq += v * v;
    }
    s1[tid] = sum; s2[tid] = sq; __syncthreads();
    for (int off = 128; off; off >>= 1) {
        if (tid < off) { s1[tid] += s1[tid + off]; s2[tid] += s2[tid + off]; }
        __syncthreads();
    }
    const float mean = s1[0] * (1.f / DMODEL);
    const float var  = s2[0] * (1.f / DMODEL) - mean * mean;
    const float inv  = rsqrtf(var + EPSV);
    #pragma unroll
    for (int i = 0; i < 4; ++i) {
        int col = tid + i * 256;
        outb[(size_t)s * DMODEL + col] =
            (__bf16)((vals[i] - mean) * inv * w[col] + b[col]);
    }
}

// ---------------------------------------------------------------------------
// f32 -> bf16 conversion with zero padding into [drows x dcols]
// ---------------------------------------------------------------------------
__global__ __launch_bounds__(256)
void k_cvt_pad(const float* __restrict__ src, __bf16* __restrict__ dst,
               int srows, int scols, int drows, int dcols)
{
    int i = blockIdx.x * 256 + threadIdx.x;
    if (i >= drows * dcols) return;
    int r = i / dcols, c = i - r * dcols;
    dst[i] = (r < srows && c < scols) ? (__bf16)src[(size_t)r * scols + c]
                                      : (__bf16)0.f;
}

// f32 -> f32 zero-padded copy (bias padding)
__global__ __launch_bounds__(256)
void k_padf(const float* __restrict__ src, float* __restrict__ dst,
            int nsrc, int ndst)
{
    int i = blockIdx.x * 256 + threadIdx.x;
    if (i < ndst) dst[i] = (i < nsrc) ? src[i] : 0.f;
}

// ---------------------------------------------------------------------------
// LDS-tiled WMMA GEMM: C[M,N] = epi(A[M,K] @ B[K,N] + bias)
// 128 threads = 4 waves; block tile 64x64x64; wave tile 16x64 (4 C tiles).
// Double-buffered async LDS staging overlaps copy with WMMA compute.
// Requires: M,N multiples of 64, K multiple of 64 (padded workspace).
// ACT: 0 none, 1 silu.  OUTB: 1 -> bf16 out, 0 -> f32 out.  RES: add residual.
// ---------------------------------------------------------------------------
template<int ACT, int OUTB, int RES>
__global__ __launch_bounds__(128)
void k_gemm(const __bf16* __restrict__ A, const __bf16* __restrict__ B,
            const float* __restrict__ bias, const float* __restrict__ resid,
            float* __restrict__ Cf, __bf16* __restrict__ Cb,
            int N, int K)
{
    __shared__ __bf16 sA[2][64][72];   // row stride 144 B (16B aligned)
    __shared__ __bf16 sB[2][64][72];
    const int tid  = threadIdx.x;
    const int lane = tid & 31, wid = tid >> 5;
    const int hi = lane >> 4, lo = lane & 15;
    const int m0 = blockIdx.y * 64, n0 = blockIdx.x * 64;

    auto stage = [&](int buf, int k0) {
        #pragma unroll
        for (int it = 0; it < 4; ++it) {
            int idx = tid + it * 128;          // 512 uint4 per matrix
            int r = idx >> 3, qd = (idx & 7) * 8;
            async_copy_b128(&sA[buf][r][qd],
                            A + (size_t)(m0 + r) * K + k0 + qd);
            async_copy_b128(&sB[buf][r][qd],
                            B + (size_t)(k0 + r) * N + n0 + qd);
        }
    };

    CFrag acc[4];
    #pragma unroll
    for (int g = 0; g < 4; ++g)
        #pragma unroll
        for (int j = 0; j < 8; ++j) acc[g].e[j] = 0.f;

    stage(0, 0);
    wait_async();
    __syncthreads();

    const int nk = K / 64;
    for (int t = 0; t < nk; ++t) {
        const int buf = t & 1;
        if (t + 1 < nk) stage(buf ^ 1, (t + 1) * 64);   // overlap with compute

        AFrag a[2];
        #pragma unroll
        for (int c = 0; c < 2; ++c)
            #pragma unroll
            for (int j = 0; j < 8; ++j)
                a[c].u[j] = *(const unsigned int*)
                    &sA[buf][wid * 16 + lo][c * 32 + a_kpair(j, hi)];

        #pragma unroll
        for (int g = 0; g < 4; ++g)
            #pragma unroll
            for (int c = 0; c < 2; ++c) {
                BFrag b;                   // B: lane<->n col, elem<->k row
                #pragma unroll
                for (int e = 0; e < 16; ++e)
                    b.e[e] = sB[buf][c * 32 + (e < 8 ? e : e + 8) + 8 * hi][g * 16 + lo];
                acc[g].v = __builtin_amdgcn_wmma_f32_16x16x32_bf16(
                    false, a[c].v, false, b.v, (short)0, acc[g].v, false, false);
            }
        wait_async();
        __syncthreads();
    }

    #pragma unroll
    for (int g = 0; g < 4; ++g)
        #pragma unroll
        for (int j = 0; j < 8; ++j) {
            int row = m0 + wid * 16 + j + 8 * hi;
            int col = n0 + g * 16 + lo;
            float v = acc[g].e[j] + bias[col];
            if (ACT == 1) v = v / (1.f + __expf(-v));           // silu
            if (RES)      v += resid[(size_t)row * N + col];
            if (OUTB) Cb[(size_t)row * N + col] = (__bf16)v;
            else      Cf[(size_t)row * N + col] = v;
        }
}

// ---------------------------------------------------------------------------
// Multi-scale retention (parallel causal form), flash style, with fused
// per-head GroupNorm + silu-gate epilogue.
// 4 waves per block; each wave owns a 16-row block of n (64 rows per block).
// K/V chunks (32 x 64) staged by double-buffered async copies shared by all
// waves.  m range is provably in [0, S_LEN): no guards (S_LEN % 32 == 0).
// ---------------------------------------------------------------------------
__global__ __launch_bounds__(128)
void k_retention(const __bf16* __restrict__ q, const __bf16* __restrict__ k,
                 const __bf16* __restrict__ v, const float* __restrict__ gate,
                 const float* __restrict__ gnw, const float* __restrict__ gnb,
                 __bf16* __restrict__ outb)
{
    const int tid  = threadIdx.x;
    const int lane = tid & 31, wid = tid >> 5;
    const int hi = lane >> 4, lo = lane & 15;
    const int h    = blockIdx.y;
    const int nblk = blockIdx.x * 64;        // block covers rows [nblk, nblk+64)
    const int n0   = nblk + wid * 16;        // this wave's 16 rows
    const float gamma = 1.f - exp2f(-5.f - (float)h);
    const float lg    = __log2f(gamma);
    const float scale = 0.125f;              // hd^-0.5, hd=64

    __shared__ __bf16 sK[2][32][72];         // K chunk, shared by 4 waves
    __shared__ __bf16 sV[2][32][72];         // V chunk, shared by 4 waves
    __shared__ __bf16 sS[4][16][32];         // per-wave score staging

    auto stageKV = [&](int buf, int m0) {
        #pragma unroll
        for (int it = 0; it < 2; ++it) {
            int idx = tid + it * 128;        // 256 uint4 per matrix
            int r = idx >> 3, qd = (idx & 7) * 8;
            async_copy_b128(&sK[buf][r][qd],
                            k + (size_t)(m0 + r) * DMODEL + h * HEADD + qd);
            async_copy_b128(&sV[buf][r][qd],
                            v + (size_t)(m0 + r) * DMODEL + h * HEADD + qd);
        }
    };

    // Q block 16x64 -> two A fragments (d chunks 0..31, 32..63)
    AFrag aq[2];
    const __bf16* qrow = q + (size_t)(n0 + lo) * DMODEL + h * HEADD;
    #pragma unroll
    for (int c = 0; c < 2; ++c)
        #pragma unroll
        for (int j = 0; j < 8; ++j)
            aq[c].u[j] = *(const unsigned int*)(qrow + c * 32 + a_kpair(j, hi));

    CFrag o[4];
    #pragma unroll
    for (int g = 0; g < 4; ++g)
        #pragma unroll
        for (int j = 0; j < 8; ++j) o[g].e[j] = 0.f;

    stageKV(0, 0);
    wait_async();
    __syncthreads();

    const int nchunks = nblk / 32 + 2;       // m0 = 0,32,... <= nblk+63
    for (int ci = 0; ci < nchunks; ++ci) {
        const int m0 = ci * 32;
        const int buf = ci & 1;
        if (ci + 1 < nchunks) stageKV(buf ^ 1, m0 + 32);   // overlap

        if (m0 <= n0 + 15) {                 // wave-uniform: EXEC stays full
            // --- scores for two 16-wide m sub-tiles ---
            #pragma unroll
            for (int t = 0; t < 2; ++t) {
                CFrag s;
                #pragma unroll
                for (int j = 0; j < 8; ++j) s.e[j] = 0.f;
                #pragma unroll
                for (int c = 0; c < 2; ++c) {
                    BFrag bk;                // B = K^T: lane<->m, elem<->d
                    bk.q[0] = *(const uint4*)&sK[buf][t * 16 + lo][c * 32 + 8 * hi];
                    bk.q[1] = *(const uint4*)&sK[buf][t * 16 + lo][c * 32 + 16 + 8 * hi];
                    s.v = __builtin_amdgcn_wmma_f32_16x16x32_bf16(
                        false, aq[c].v, false, bk.v, (short)0, s.v, false, false);
                }
                // decay + causal mask + scale, C layout -> LDS staging
                #pragma unroll
                for (int j = 0; j < 8; ++j) {
                    int n = n0 + j + 8 * hi;
                    int m = m0 + t * 16 + lo;
                    float f = (m <= n) ? scale * exp2f(lg * (float)(n - m)) : 0.f;
                    sS[wid][j + 8 * hi][t * 16 + lo] = (__bf16)(s.e[j] * f);
                }
            }

            // re-stripe score tile (16x32) into A-operand layout (same wave:
            // LDS ops are in-order, no barrier needed)
            AFrag as;
            #pragma unroll
            for (int j = 0; j < 8; ++j)
                as.u[j] = *(const unsigned int*)&sS[wid][lo][a_kpair(j, hi)];

            // out += S_tile @ V_chunk (32 x 64)
            #pragma unroll
            for (int g = 0; g < 4; ++g) {
                BFrag bv;                    // B = V: lane<->col, elem<->m row
                #pragma unroll
                for (int e = 0; e < 16; ++e)
                    bv.e[e] = sV[buf][(e < 8 ? e : e + 8) + 8 * hi][g * 16 + lo];
                o[g].v = __builtin_amdgcn_wmma_f32_16x16x32_bf16(
                    false, as.v, false, bv.v, (short)0, o[g].v, false, false);
            }
        }
        wait_async();
        __syncthreads();                     // next chunk may reuse buffers
    }

    // ---- fused GroupNorm (over hd=64) + silu-gate, bf16 output ----
    // Row r = j + 8*hi lives in one 16-lane half; reduce with xor shuffles.
    #pragma unroll
    for (int j = 0; j < 8; ++j) {
        float p  = o[0].e[j] + o[1].e[j] + o[2].e[j] + o[3].e[j];
        float p2 = o[0].e[j] * o[0].e[j] + o[1].e[j] * o[1].e[j]
                 + o[2].e[j] * o[2].e[j] + o[3].e[j] * o[3].e[j];
        #pragma unroll
        for (int mask = 1; mask <= 8; mask <<= 1) {
            p  += __shfl_xor(p,  mask);
            p2 += __shfl_xor(p2, mask);
        }
        const float mean = p * (1.f / HEADD);
        const float inv  = rsqrtf(p2 * (1.f / HEADD) - mean * mean + EPSV);
        const int row = n0 + j + 8 * hi;
        #pragma unroll
        for (int g = 0; g < 4; ++g) {
            const int col = h * HEADD + g * 16 + lo;
            float val = (o[g].e[j] - mean) * inv * gnw[col] + gnb[col];
            val *= gate[(size_t)row * DMODEL + col];
            outb[(size_t)row * DMODEL + col] = (__bf16)val;
        }
    }
}

// ---------------------------------------------------------------------------
// SwiGLU elementwise: h = silu(u) * g, zero-padded bf16 [S_LEN x HPAD]
// uv is padded [S_LEN x NINPAD]; u = cols [0,HIDDEN), g = cols [HIDDEN,NIN)
// ---------------------------------------------------------------------------
__global__ __launch_bounds__(256)
void k_silu_mul(const float* __restrict__ uv, __bf16* __restrict__ out, int n)
{
    int i = blockIdx.x * 256 + threadIdx.x;
    if (i >= n) return;
    int r = i / HPAD, c = i - r * HPAD;
    float val = 0.f;
    if (c < HIDDEN) {
        float u = uv[(size_t)r * NINPAD + c];
        float g = uv[(size_t)r * NINPAD + HIDDEN + c];
        val = (u / (1.f + __expf(-u))) * g;
    }
    out[i] = (__bf16)val;
}

// ---------------------------------------------------------------------------
// Host-side orchestration
// ---------------------------------------------------------------------------
extern "C" void kernel_launch(void* const* d_in, const int* in_sizes, int n_in,
                              void* d_out, int out_size, void* d_ws, size_t ws_size,
                              hipStream_t stream)
{
    (void)in_sizes; (void)n_in; (void)out_size; (void)ws_size;
    const float* x   = (const float*)d_in[0];
    const float* n1w = (const float*)d_in[1];
    const float* n1b = (const float*)d_in[2];
    const float* wq  = (const float*)d_in[3];  const float* bq = (const float*)d_in[4];
    const float* wk  = (const float*)d_in[5];  const float* bk = (const float*)d_in[6];
    const float* wv  = (const float*)d_in[7];  const float* bv = (const float*)d_in[8];
    const float* wg  = (const float*)d_in[9];  const float* bg = (const float*)d_in[10];
    const float* wo  = (const float*)d_in[11]; const float* bo = (const float*)d_in[12];
    const float* gnw = (const float*)d_in[13]; const float* gnb = (const float*)d_in[14];
    const float* n2w = (const float*)d_in[15]; const float* n2b = (const float*)d_in[16];
    const float* w_in  = (const float*)d_in[17]; const float* b_in  = (const float*)d_in[18];
    const float* w_out = (const float*)d_in[19]; const float* b_out = (const float*)d_in[20];
    float* out = (float*)d_out;

    char* ws = (char*)d_ws;
    size_t off = 0;
    auto take = [&](size_t bytes) -> void* {
        void* p = ws + off;
        off = (off + bytes + 255) & ~(size_t)255;
        return p;
    };
    const size_t SD = (size_t)S_LEN * DMODEL;

    __bf16* xnb  = (__bf16*)take(SD * 2);
    __bf16* wqb  = (__bf16*)take((size_t)DMODEL * DMODEL * 2);
    __bf16* wkb  = (__bf16*)take((size_t)DMODEL * DMODEL * 2);
    __bf16* wvb  = (__bf16*)take((size_t)DMODEL * DMODEL * 2);
    __bf16* wgb  = (__bf16*)take((size_t)DMODEL * DMODEL * 2);
    __bf16* wob  = (__bf16*)take((size_t)DMODEL * DMODEL * 2);
    __bf16* winb = (__bf16*)take((size_t)DMODEL * NINPAD * 2);   // padded N
    __bf16* woutb= (__bf16*)take((size_t)HPAD * DMODEL * 2);     // padded K rows
    float*  binp = (float*) take((size_t)NINPAD * 4);            // padded bias
    __bf16* qb   = (__bf16*)take(SD * 2);
    __bf16* kb   = (__bf16*)take(SD * 2);
    __bf16* vb   = (__bf16*)take(SD * 2);
    float*  gatef= (float*) take(SD * 4);
    __bf16* a2b  = (__bf16*)take(SD * 2);
    float*  x1f  = (float*) take(SD * 4);
    __bf16* x2b  = (__bf16*)take(SD * 2);
    float*  uvf  = (float*) take((size_t)S_LEN * NINPAD * 4);    // padded
    __bf16* h1b  = (__bf16*)take((size_t)S_LEN * HPAD * 2);      // padded

    // 1. LN1 -> bf16
    k_ln<<<S_LEN, 256, 0, stream>>>(x, n1w, n1b, xnb);

    // 2. weight conversions (with zero padding to tile multiples)
    auto cvt = [&](const float* src, __bf16* dst, int sr, int sc, int dr, int dc) {
        int n = dr * dc;
        k_cvt_pad<<<(n + 255) / 256, 256, 0, stream>>>(src, dst, sr, sc, dr, dc);
    };
    cvt(wq, wqb, DMODEL, DMODEL, DMODEL, DMODEL);
    cvt(wk, wkb, DMODEL, DMODEL, DMODEL, DMODEL);
    cvt(wv, wvb, DMODEL, DMODEL, DMODEL, DMODEL);
    cvt(wg, wgb, DMODEL, DMODEL, DMODEL, DMODEL);
    cvt(wo, wob, DMODEL, DMODEL, DMODEL, DMODEL);
    cvt(w_in,  winb,  DMODEL, NIN,  DMODEL, NINPAD);
    cvt(w_out, woutb, HIDDEN, DMODEL, HPAD, DMODEL);
    k_padf<<<(NINPAD + 255) / 256, 256, 0, stream>>>(b_in, binp, NIN, NINPAD);

    const dim3 blk(128);
    const dim3 gD(DMODEL / 64, S_LEN / 64);          // 16 x 32
    const dim3 gIn(NINPAD / 64, S_LEN / 64);         // 86 x 32

    // 3. projections: q, k, v (bf16 out), gate (f32 + silu)
    k_gemm<0,1,0><<<gD, blk, 0, stream>>>(xnb, wqb, bq, nullptr, nullptr, qb,
                                          DMODEL, DMODEL);
    k_gemm<0,1,0><<<gD, blk, 0, stream>>>(xnb, wkb, bk, nullptr, nullptr, kb,
                                          DMODEL, DMODEL);
    k_gemm<0,1,0><<<gD, blk, 0, stream>>>(xnb, wvb, bv, nullptr, nullptr, vb,
                                          DMODEL, DMODEL);
    k_gemm<1,0,0><<<gD, blk, 0, stream>>>(xnb, wgb, bg, nullptr, gatef, nullptr,
                                          DMODEL, DMODEL);

    // 4. retention + fused GroupNorm/gate -> bf16 A for output projection
    k_retention<<<dim3(S_LEN / 64, NHEADS), 128, 0, stream>>>(
        qb, kb, vb, gatef, gnw, gnb, a2b);

    // 5. output projection + residual 1
    k_gemm<0,0,1><<<gD, blk, 0, stream>>>(a2b, wob, bo, x, x1f, nullptr,
                                          DMODEL, DMODEL);

    // 6. LN2 -> bf16
    k_ln<<<S_LEN, 256, 0, stream>>>(x1f, n2w, n2b, x2b);

    // 7. FFN in (padded N = 5504)
    k_gemm<0,0,0><<<gIn, blk, 0, stream>>>(x2b, winb, binp, nullptr, uvf, nullptr,
                                           NINPAD, DMODEL);

    // 8. SwiGLU elementwise -> zero-padded [S_LEN x HPAD]
    {
        int n = S_LEN * HPAD;
        k_silu_mul<<<(n + 255) / 256, 256, 0, stream>>>(uvf, h1b, n);
    }

    // 9. FFN out (padded K = 2752) + residual 2 -> d_out
    k_gemm<0,0,1><<<gD, blk, 0, stream>>>(h1b, woutb, b_out, x1f, out, nullptr,
                                          DMODEL, HPAD);
}
